// TemporalGNN_82325933130191
// MI455X (gfx1250) — matrix operand order repaired and checked by reading, not compiled, not run on hardware
//
#include <hip/hip_runtime.h>

typedef __bf16 bf16_t;
typedef __bf16 v16bf __attribute__((ext_vector_type(16)));
typedef __bf16 bf16x8 __attribute__((ext_vector_type(8)));
typedef float  v8f   __attribute__((ext_vector_type(8)));
typedef _Float16 f16x2 __attribute__((ext_vector_type(2)));
typedef _Float16 f16x8 __attribute__((ext_vector_type(8)));

#define HDIM 256
#define GDIM 1024   // 4*H gates
#define CDIM 8

// ---------------------------------------------------------------------------
// Elementwise / scatter helpers
// ---------------------------------------------------------------------------
__global__ void zero_f32(float* p, long n) {
    long i = (long)blockIdx.x * blockDim.x + threadIdx.x;
    if (i < n) p[i] = 0.0f;
}

__global__ void deg_count(const int* dst, int E, float* deg) {
    int i = blockIdx.x * blockDim.x + threadIdx.x;
    if (i < E) atomicAdd(&deg[dst[i]], 1.0f);
}

// deg includes the self-loop (+1); reference deg>0 always holds then.
__global__ void deg_finalize(float* deg, int n) {
    int i = blockIdx.x * blockDim.x + threadIdx.x;
    if (i < n) deg[i] = rsqrtf(deg[i] + 1.0f);
}

// f32 -> bf16 (optionally fused relu), row-major, vector4 granularity
__global__ void cvt_bf16(const float* __restrict__ in, bf16_t* __restrict__ out,
                         long n4, int relu) {
    long i = (long)blockIdx.x * blockDim.x + threadIdx.x;
    if (i >= n4) return;
    const float4 v = *(const float4*)(in + 4 * i);
    float a = v.x, b = v.y, c = v.z, d = v.w;
    if (relu) { a = fmaxf(a, 0.f); b = fmaxf(b, 0.f); c = fmaxf(c, 0.f); d = fmaxf(d, 0.f); }
    bf16_t* o = out + 4 * i;
    o[0] = (bf16_t)a; o[1] = (bf16_t)b; o[2] = (bf16_t)c; o[3] = (bf16_t)d;
}

// Pack B (f32) into bf16 WMMA 32x16 B-fragment layout:
//   Bp[((tn*kchunks + kc)*32 + lane)*16 + e]  <-  B[k, n]
//   n = tn*16 + (lane&15),  k = kc*32 + (lane>=16 ? 16 : 0) + e
// transB=1: source is [Nc x K] row-major (multiply by B^T).
__global__ void pack_b(const float* __restrict__ B, bf16_t* __restrict__ Bp,
                       int K, int Nc, int transB) {
    long idx = (long)blockIdx.x * blockDim.x + threadIdx.x;
    if (idx >= (long)K * Nc) return;
    const int e    = (int)(idx & 15);
    const int lane = (int)((idx >> 4) & 31);
    const long rest = idx >> 9;
    const int kchunks = K >> 5;
    const int kc = (int)(rest % kchunks);
    const int tn = (int)(rest / kchunks);
    const int n = tn * 16 + (lane & 15);
    const int k = kc * 32 + ((lane >= 16) ? 16 : 0) + e;
    const float v = transB ? B[(long)n * K + k] : B[(long)k * Nc + n];
    Bp[idx] = (bf16_t)v;
}

// out[i,c] = bias[c] + dinv[i]^2 * XW[i,c]   (self-loop term + bias preset)
__global__ void gcn_init(const float* __restrict__ XW, const float* __restrict__ dinv,
                         const float* __restrict__ bias, float* __restrict__ out, int n) {
    long idx = (long)blockIdx.x * blockDim.x + threadIdx.x;
    if (idx >= (long)n * HDIM) return;
    int i = (int)(idx >> 8);
    int c = (int)(idx & (HDIM - 1));
    float dv = dinv[i];
    out[idx] = bias[c] + dv * dv * XW[idx];
}

// out[dst,:] += dinv[src]*dinv[dst] * XW[src,:]  (64 lanes/edge, float4 per lane)
__global__ void gcn_scatter(const float* __restrict__ XW, const float* __restrict__ dinv,
                            const int* __restrict__ src, const int* __restrict__ dst,
                            int E, float* __restrict__ out) {
    long gid = (long)blockIdx.x * blockDim.x + threadIdx.x;
    if (gid >= (long)E * 64) return;
    int e  = (int)(gid >> 6);
    int c4 = (int)(gid & 63) << 2;
    int s = src[e], d = dst[e];
    float nrm = dinv[s] * dinv[d];
    const float4 v = *(const float4*)(XW + (long)s * HDIM + c4);
    float* o = out + (long)d * HDIM + c4;
    atomicAdd(o + 0, nrm * v.x);
    atomicAdd(o + 1, nrm * v.y);
    atomicAdd(o + 2, nrm * v.z);
    atomicAdd(o + 3, nrm * v.w);
}

__global__ void bias_sum(const float* a, const float* b, float* o, int n) {
    int i = blockIdx.x * blockDim.x + threadIdx.x;
    if (i < n) o[i] = a[i] + b[i];
}

// ---------------------------------------------------------------------------
// bf16 WMMA GEMM:  D[M x Nc] = A[M x K] * B (+bias per column)
//   A  : bf16 row-major  (pre-converted)
//   Bp : bf16 pre-packed 32x16 wave fragments (see pack_b)
// Inner loop = 4x global_load_b128 + 1x v_wmma_f32_16x16x32_bf16.
// 8 waves/block cover consecutive M tiles of one N tile (B fragment reuse).
// ---------------------------------------------------------------------------
__global__ void __launch_bounds__(256)
gemm_bf16_wmma(const bf16_t* __restrict__ A, const bf16_t* __restrict__ Bp,
               float* __restrict__ D, int M, int K, int Nc,
               const float* __restrict__ bias) {
    const int wave = threadIdx.x >> 5;
    const int lane = threadIdx.x & 31;
    const bool hi  = lane >= 16;
    const int l16  = lane & 15;

    const int Mt = (M + 15) >> 4;
    const int Nt = Nc >> 4;
    long tile = (long)blockIdx.x * (blockDim.x >> 5) + wave;
    if (tile >= (long)Mt * Nt) return;
    const int tm = (int)(tile % Mt);          // consecutive waves -> consecutive M tiles
    const int tn = (int)(tile / Mt);          // ... sharing one packed-B panel

    int row = tm * 16 + l16;
    if (row >= M) row = M - 1;                // clamp (store is guarded)

    const int kchunks = K >> 5;
    const bf16_t* ap = A + (long)row * K + (hi ? 16 : 0);
    const bf16_t* bp = Bp + (long)tn * kchunks * 512 + lane * 16;

    v8f acc = {};
    for (int kc = 0; kc < kchunks; ++kc) {
        const bf16x8 a0 = *(const bf16x8*)(ap + 0);
        const bf16x8 a1 = *(const bf16x8*)(ap + 8);
        const bf16x8 b0 = *(const bf16x8*)(bp + 0);
        const bf16x8 b1 = *(const bf16x8*)(bp + 8);
        const v16bf a = __builtin_shufflevector(a0, a1, 0,1,2,3,4,5,6,7,8,9,10,11,12,13,14,15);
        const v16bf b = __builtin_shufflevector(b0, b1, 0,1,2,3,4,5,6,7,8,9,10,11,12,13,14,15);
        acc = __builtin_amdgcn_wmma_f32_16x16x32_bf16(
            /*neg_a=*/false, a, /*neg_b=*/false, b,
            /*c_mod=*/(short)0, acc, /*reuse_a=*/false, /*reuse_b=*/false);
        ap += 32;
        bp += 512;
    }

    const int col = tn * 16 + l16;
    const float bb = bias ? bias[col] : 0.0f;
#pragma unroll
    for (int r = 0; r < 8; ++r) {
        const int crow = tm * 16 + r + (hi ? 8 : 0);
        if (crow < M) D[(long)crow * Nc + col] = acc[r] + bb;
    }
}

// ---------------------------------------------------------------------------
// Persistent single-workgroup LSTM recurrence.
// 1024 threads = 32 waves on one WGP; thread t owns gate row t of w_hh,
// weights pinned in 128 packed-fp16 VGPRs; hidden state broadcast via LDS
// (read as f16x8 -> ds_load_b128); dots via v_dot2_f32_f16.
// G holds precomputed x@w_ih^T + (b_ih+b_hh).
// ---------------------------------------------------------------------------
__global__ void __launch_bounds__(1024, 1)
lstm_recurrent(const float* __restrict__ G, const float* __restrict__ w_hh,
               int T, float* __restrict__ hT_out) {
    __shared__ __align__(16) _Float16 hbuf[HDIM];
    __shared__ float gates[GDIM];

    const int t = threadIdx.x;                 // gate index 0..1023

    // Pin row t of w_hh (256 fp32 -> 128 packed fp16 dwords) in registers.
    f16x2 w[HDIM / 2];
#pragma unroll
    for (int j = 0; j < HDIM / 2; ++j) {
        const float2 f = *(const float2*)(w_hh + (long)t * HDIM + 2 * j);
        f16x2 p; p[0] = (_Float16)f.x; p[1] = (_Float16)f.y;
        w[j] = p;
    }

    float c = 0.0f;                            // cell state (threads 0..255)
    if (t < HDIM) hbuf[t] = (_Float16)0.0f;
    __syncthreads();

    for (int step = 0; step < T; ++step) {
        float acc = G[(long)step * GDIM + t];  // precomputed x-term + biases
        if (step + 1 < T)
            __builtin_prefetch(G + (long)(step + 1) * GDIM + t, 0, 1);

#pragma unroll
        for (int j = 0; j < HDIM / 8; ++j) {   // 32 x ds_load_b128
            const f16x8 hv = *(const f16x8*)(&hbuf[8 * j]);
            const f16x2 h0 = __builtin_shufflevector(hv, hv, 0, 1);
            const f16x2 h1 = __builtin_shufflevector(hv, hv, 2, 3);
            const f16x2 h2 = __builtin_shufflevector(hv, hv, 4, 5);
            const f16x2 h3 = __builtin_shufflevector(hv, hv, 6, 7);
#if __has_builtin(__builtin_amdgcn_fdot2)
            acc = __builtin_amdgcn_fdot2(w[4 * j + 0], h0, acc, false);
            acc = __builtin_amdgcn_fdot2(w[4 * j + 1], h1, acc, false);
            acc = __builtin_amdgcn_fdot2(w[4 * j + 2], h2, acc, false);
            acc = __builtin_amdgcn_fdot2(w[4 * j + 3], h3, acc, false);
#else
            acc = fmaf((float)w[4*j+0][0], (float)h0[0], acc);
            acc = fmaf((float)w[4*j+0][1], (float)h0[1], acc);
            acc = fmaf((float)w[4*j+1][0], (float)h1[0], acc);
            acc = fmaf((float)w[4*j+1][1], (float)h1[1], acc);
            acc = fmaf((float)w[4*j+2][0], (float)h2[0], acc);
            acc = fmaf((float)w[4*j+2][1], (float)h2[1], acc);
            acc = fmaf((float)w[4*j+3][0], (float)h3[0], acc);
            acc = fmaf((float)w[4*j+3][1], (float)h3[1], acc);
#endif
        }
        gates[t] = acc;
        __syncthreads();

        if (t < HDIM) {
            const float ig = gates[t];
            const float fg = gates[HDIM + t];
            const float gg = gates[2 * HDIM + t];
            const float og = gates[3 * HDIM + t];
            const float si = 1.0f / (1.0f + expf(-ig));
            const float sf = 1.0f / (1.0f + expf(-fg));
            const float so = 1.0f / (1.0f + expf(-og));
            c = sf * c + si * tanhf(gg);
            const float hv = so * tanhf(c);
            hbuf[t] = (_Float16)hv;
            if (step == T - 1) hT_out[t] = hv;
        }
        __syncthreads();
    }
}

// out[c] = bc[c] + sum_k hT[k] * Wc[k, c]
__global__ void classify(const float* __restrict__ hT, const float* __restrict__ Wc,
                         const float* __restrict__ bc, float* __restrict__ out) {
    const int c = threadIdx.x;
    if (c < CDIM) {
        float a = bc[c];
        for (int k = 0; k < HDIM; ++k) a = fmaf(hT[k], Wc[k * CDIM + c], a);
        out[c] = a;
    }
}

// ---------------------------------------------------------------------------
extern "C" void kernel_launch(void* const* d_in, const int* in_sizes, int n_in,
                              void* d_out, int out_size, void* d_ws, size_t ws_size,
                              hipStream_t stream) {
    const float* x     = (const float*)d_in[0];
    const int*   eidx  = (const int*)d_in[1];
    const float* W1    = (const float*)d_in[3];
    const float* b1    = (const float*)d_in[4];
    const float* W2    = (const float*)d_in[5];
    const float* b2    = (const float*)d_in[6];
    const float* w_ih  = (const float*)d_in[7];
    const float* w_hh  = (const float*)d_in[8];
    const float* b_ih  = (const float*)d_in[9];
    const float* b_hh  = (const float*)d_in[10];
    const float* Wc    = (const float*)d_in[11];
    const float* bc    = (const float*)d_in[12];

    const int N = in_sizes[0] / HDIM;
    const int E = in_sizes[1] / 2;
    const int* src = eidx;
    const int* dst = eidx + E;

    // ---- workspace layout
    char* wsp = (char*)d_ws;
    size_t off = 0;
    auto alloc = [&](size_t bytes) -> char* {
        char* p = wsp + off;
        off += (bytes + 255) & ~(size_t)255;
        return p;
    };
    float*  dinv = (float*)alloc((size_t)N * 4);
    float*  bufA = (float*)alloc((size_t)N * HDIM * 4);   // GEMM outputs (XW)
    float*  bufB = (float*)alloc((size_t)N * HDIM * 4);   // aggregated h
    float*  bufG = (float*)alloc((size_t)N * GDIM * 4);   // LSTM gate pre-activations
    bf16_t* Abf  = (bf16_t*)alloc((size_t)N * HDIM * 2);  // bf16 A operand (reused)
    bf16_t* Bpk  = (bf16_t*)alloc((size_t)HDIM * GDIM * 2); // packed B (reused)
    float*  bsum = (float*)alloc(GDIM * 4);
    float*  hT   = (float*)alloc(HDIM * 4);
    (void)ws_size;

    const int TB = 256;
    auto grid1 = [&](long n) { return (unsigned)((n + TB - 1) / TB); };

    // ---- degree / normalization
    zero_f32<<<grid1(N), TB, 0, stream>>>(dinv, N);
    deg_count<<<grid1(E), TB, 0, stream>>>(dst, E, dinv);
    deg_finalize<<<grid1(N), TB, 0, stream>>>(dinv, N);

    const long tilesH = (long)((N + 15) / 16) * (HDIM / 16);
    const long tilesG = (long)((N + 15) / 16) * (GDIM / 16);
    const unsigned gemmGridH = (unsigned)((tilesH + 7) / 8);
    const unsigned gemmGridG = (unsigned)((tilesG + 7) / 8);
    const long nElemH = (long)N * HDIM;

    // ---- GCN layer 1: XW1 = x @ W1
    cvt_bf16<<<grid1(nElemH / 4), TB, 0, stream>>>(x, Abf, nElemH / 4, 0);
    pack_b<<<grid1((long)HDIM * HDIM), TB, 0, stream>>>(W1, Bpk, HDIM, HDIM, 0);
    gemm_bf16_wmma<<<gemmGridH, TB, 0, stream>>>(Abf, Bpk, bufA, N, HDIM, HDIM, nullptr);
    gcn_init<<<grid1(nElemH), TB, 0, stream>>>(bufA, dinv, b1, bufB, N);
    gcn_scatter<<<grid1((long)E * 64), TB, 0, stream>>>(bufA, dinv, src, dst, E, bufB);

    // ---- GCN layer 2: XW2 = relu(h1) @ W2
    cvt_bf16<<<grid1(nElemH / 4), TB, 0, stream>>>(bufB, Abf, nElemH / 4, 1);
    pack_b<<<grid1((long)HDIM * HDIM), TB, 0, stream>>>(W2, Bpk, HDIM, HDIM, 0);
    gemm_bf16_wmma<<<gemmGridH, TB, 0, stream>>>(Abf, Bpk, bufA, N, HDIM, HDIM, nullptr);
    gcn_init<<<grid1(nElemH), TB, 0, stream>>>(bufA, dinv, b2, bufB, N);
    gcn_scatter<<<grid1((long)E * 64), TB, 0, stream>>>(bufA, dinv, src, dst, E, bufB);

    // ---- LSTM input projection: G = relu(h2) @ w_ih^T + (b_ih + b_hh)
    cvt_bf16<<<grid1(nElemH / 4), TB, 0, stream>>>(bufB, Abf, nElemH / 4, 1);
    bias_sum<<<grid1(GDIM), TB, 0, stream>>>(b_ih, b_hh, bsum, GDIM);
    pack_b<<<grid1((long)HDIM * GDIM), TB, 0, stream>>>(w_ih, Bpk, HDIM, GDIM, 1);
    gemm_bf16_wmma<<<gemmGridG, TB, 0, stream>>>(Abf, Bpk, bufG, N, HDIM, GDIM, bsum);

    // ---- sequential LSTM recurrence (single persistent workgroup)
    lstm_recurrent<<<1, 1024, 0, stream>>>(bufG, w_hh, N, hT);

    // ---- classifier
    classify<<<1, 32, 0, stream>>>(hT, Wc, bc, (float*)d_out);
}
